// FilteringModel_55095840473282
// MI455X (gfx1250) — matrix operand order repaired and checked by reading, not compiled, not run on hardware
//
#include <hip/hip_runtime.h>

typedef __attribute__((ext_vector_type(16))) _Float16     v16h;
typedef __attribute__((ext_vector_type(8)))  float        v8f;
typedef __attribute__((ext_vector_type(8)))  unsigned int v8u;

#define HD      128   // hidden width (H1 == H2 == 128)
#define FIN     32    // 2*F concat input width
#define PITCH1  40    // LDS row pitch (halfs) for W1^T : 32 + 8 pad (bank-conflict free)
#define PITCH2  136   // LDS row pitch (halfs) for W2^T : 128 + 8 pad
#define BLOCK   256   // 8 wave32 per block
#define BN_EPS  1e-5f

__device__ __forceinline__ unsigned short h16(float f) {
  return __builtin_bit_cast(unsigned short, (_Float16)f);
}
__device__ __forceinline__ unsigned int pk2(float a, float b) {
  return (unsigned int)h16(a) | ((unsigned int)h16(b) << 16);
}
// A-fragment load: k order used here (B matches) = per lane K[kb+2*off8 .. +15] (contiguous)
__device__ __forceinline__ v16h loadA(const unsigned short* base, int row, int pitch,
                                      int kbase, int off8) {
  const unsigned short* p = base + row * pitch + kbase + 2 * off8;
  uint4 lo = *(const uint4*)(p);
  uint4 hi = *(const uint4*)(p + 8);
  v8u r;
  r[0] = lo.x; r[1] = lo.y; r[2] = lo.z; r[3] = lo.w;
  r[4] = hi.x; r[5] = hi.y; r[6] = hi.z; r[7] = hi.w;
  return __builtin_bit_cast(v16h, r);
}
__device__ __forceinline__ v8f wmma16(v16h a, v16h b, v8f c) {
  return __builtin_amdgcn_wmma_f32_16x16x32_f16(false, a, false, b, (short)0, c, false, false);
}
__device__ __forceinline__ unsigned int sx16(unsigned int v) {
  return (unsigned int)__shfl_xor((int)v, 16, 32);
}

__global__ void zero_stats_kernel(float* ws) {
  int i = threadIdx.x;
  if (i < 512) ws[i] = 0.0f;
}

// coef[f] = (a, c, w3, 0) with  h = relu(z*a + c);  bias absorbs into BN (training mode)
__global__ void finalize_kernel(const float* __restrict__ s, const float* __restrict__ q,
                                const float* __restrict__ gamma, const float* __restrict__ beta,
                                const float* __restrict__ w3, float4* __restrict__ coef,
                                float invE) {
  int f = threadIdx.x;
  if (f >= HD) return;
  float mu  = s[f] * invE;
  float var = fmaxf(q[f] * invE - mu * mu, 0.0f);
  float a   = gamma[f] * rsqrtf(var + BN_EPS);
  float c   = beta[f] - mu * a;
  float w   = w3 ? w3[f] : 0.0f;
  coef[f] = make_float4(a, c, w, 0.0f);
}

// PHASE 1: z1 stats.  PHASE 2: h1 -> z2 stats.  PHASE 3: full forward -> out.
// TP = edge tiles (of 16) processed per wave iteration.
template <int PHASE>
__global__ void __launch_bounds__(BLOCK)
edge_mlp_kernel(const float* __restrict__ x, const int* __restrict__ ei,
                const float* __restrict__ W1g, const float* __restrict__ W2g,
                const float4* __restrict__ coef1, const float4* __restrict__ coef2,
                float* __restrict__ gsum, float* __restrict__ gsq,
                float* __restrict__ out, const float* __restrict__ b3p, int E) {
  constexpr int TP = (PHASE == 2) ? 1 : 2;

  __shared__ __align__(16) unsigned short sW1[HD * PITCH1];
  __shared__ __align__(16) unsigned short sW2[(PHASE >= 2) ? HD * PITCH2 : 1];
  __shared__ float4 sC1[(PHASE >= 2) ? HD : 1];
  __shared__ float4 sC2[(PHASE == 3) ? HD : 1];
  __shared__ float  sSum[(PHASE <= 2) ? HD : 1];
  __shared__ float  sSq [(PHASE <= 2) ? HD : 1];

  const int tid = threadIdx.x;

  // ---- prologue: stage W^T (fp16) + BN coefficients into LDS ----
  for (int i = tid; i < FIN * HD; i += BLOCK) {          // sW1[m][k] = W1[k][m]
    int k = i >> 7, m = i & 127;
    sW1[m * PITCH1 + k] = h16(W1g[i]);
  }
  if (PHASE >= 2) {
    for (int i = tid; i < HD * HD; i += BLOCK) {         // sW2[m][k] = W2[k][m]
      int k = i >> 7, m = i & 127;
      sW2[m * PITCH2 + k] = h16(W2g[i]);
    }
    for (int i = tid; i < HD; i += BLOCK) sC1[i] = coef1[i];
  }
  if (PHASE == 3) {
    for (int i = tid; i < HD; i += BLOCK) sC2[i] = coef2[i];
  }
  if (PHASE <= 2) {
    for (int i = tid; i < HD; i += BLOCK) { sSum[i] = 0.0f; sSq[i] = 0.0f; }
  }
  __syncthreads();

  const int lane = tid & 31;
  const int half = lane >> 4;        // 0: K/feat low half, 1: high half
  const int eL   = lane & 15;        // edge slot within tile / A-matrix M row
  const int off8 = half * 8;
  const int wid  = tid >> 5;
  const int gw   = blockIdx.x * (BLOCK >> 5) + wid;
  const int nwav = gridDim.x * (BLOCK >> 5);
  const int step = TP * 16;
  const int nit  = (E + step - 1) / step;
  const float b3v = (PHASE == 3) ? b3p[0] : 0.0f;

  float S[8][8], Q[8][8];
  if (PHASE <= 2) {
#pragma unroll
    for (int m = 0; m < 8; ++m)
#pragma unroll
      for (int r = 0; r < 8; ++r) { S[m][r] = 0.0f; Q[m][r] = 0.0f; }
  }

  for (int it = gw; it < nit; it += nwav) {
    int   e[TP];
    float vm[TP];
    v16h  B1[TP];
#pragma unroll
    for (int t = 0; t < TP; ++t) {
      e[t] = it * step + t * 16 + eL;
      const bool valid = (e[t] < E);
      vm[t] = valid ? 1.0f : 0.0f;
      // lanes 0-15: row node (K=0..15); lanes 16-31: col node (K=16..31)
      int node = 0;
      if (valid) node = (half == 0) ? ei[e[t]] : ei[E + e[t]];
      const float4* xp = (const float4*)(x + (size_t)node * 16);
      float4 f0 = xp[0], f1 = xp[1], f2 = xp[2], f3 = xp[3];
      v8u bp;  // B1: 32x16 f16, N=edge in lanes, K split across wave halves
      bp[0] = pk2(f0.x, f0.y); bp[1] = pk2(f0.z, f0.w);
      bp[2] = pk2(f1.x, f1.y); bp[3] = pk2(f1.z, f1.w);
      bp[4] = pk2(f2.x, f2.y); bp[5] = pk2(f2.z, f2.w);
      bp[6] = pk2(f3.x, f3.y); bp[7] = pk2(f3.z, f3.w);
      B1[t] = __builtin_bit_cast(v16h, bp);
    }

    // ---- layer 1 GEMM, software-pipelined over m (ping-pong D, no hazard NOPs) ----
    unsigned int P1[TP][8][4];   // packed fp16 h1 (phases 2,3)
    v8f D[2][TP];

    auto consume_l1 = [&](int m, v8f (&d)[TP]) {
#pragma unroll
      for (int t = 0; t < TP; ++t) {
        if (PHASE == 1) {
#pragma unroll
          for (int r = 0; r < 8; ++r) {
            S[m][r] += vm[t] * d[t][r];
            Q[m][r] += vm[t] * d[t][r] * d[t][r];
          }
        } else {
#pragma unroll
          for (int r = 0; r < 8; ++r) {
            float4 cf = sC1[16 * m + r + off8];
            d[t][r] = fmaxf(d[t][r] * cf.x + cf.y, 0.0f);
          }
          P1[t][m][0] = pk2(d[t][0], d[t][1]); P1[t][m][1] = pk2(d[t][2], d[t][3]);
          P1[t][m][2] = pk2(d[t][4], d[t][5]); P1[t][m][3] = pk2(d[t][6], d[t][7]);
        }
      }
    };

#pragma unroll
    for (int m = 0; m < 8; ++m) {
      const int cur = m & 1;
      v16h A = loadA(sW1, 16 * m + eL, PITCH1, 0, off8);
#pragma unroll
      for (int t = 0; t < TP; ++t) {
        v8f z = {};
        D[cur][t] = wmma16(A, B1[t], z);
      }
      if (m > 0) consume_l1(m - 1, D[cur ^ 1]);
    }
    consume_l1(7, D[1]);

    if (PHASE >= 2) {
      // ---- h1^T as B fragments for K-steps j=0..3 (half-wave swap via shfl_xor 16) ----
      v16h B2[TP][4];
#pragma unroll
      for (int t = 0; t < TP; ++t)
#pragma unroll
        for (int j = 0; j < 4; ++j) {
          v8u b2p;
#pragma unroll
          for (int v = 0; v < 4; ++v) {
            unsigned int hiv = sx16(P1[t][2 * j + 1][v]); // lo-half of mtile 2j+1 -> upper lanes
            unsigned int lov = sx16(P1[t][2 * j][v]);     // hi-half of mtile 2j   -> lower lanes
            b2p[v]     = (half == 0) ? P1[t][2 * j][v] : hiv;
            b2p[v + 4] = (half == 0) ? lov : P1[t][2 * j + 1][v];
          }
          B2[t][j] = __builtin_bit_cast(v16h, b2p);
        }

      // ---- layer 2 GEMM (K=128 -> 4 wmma per m-tile), pipelined over m ----
      float dot[TP];
#pragma unroll
      for (int t = 0; t < TP; ++t) dot[t] = 0.0f;
      v8f acc[2][TP];

      auto consume_l2 = [&](int m, v8f (&a)[TP]) {
#pragma unroll
        for (int t = 0; t < TP; ++t) {
          if (PHASE == 2) {
#pragma unroll
            for (int r = 0; r < 8; ++r) {
              S[m][r] += vm[t] * a[t][r];
              Q[m][r] += vm[t] * a[t][r] * a[t][r];
            }
          } else {
#pragma unroll
            for (int r = 0; r < 8; ++r) {
              float4 cf = sC2[16 * m + r + off8];
              float h = fmaxf(a[t][r] * cf.x + cf.y, 0.0f);
              dot[t] += h * cf.z;                        // cf.z = W3[f]
            }
          }
        }
      };

#pragma unroll
      for (int m = 0; m < 8; ++m) {
        const int cur = m & 1;
#pragma unroll
        for (int j = 0; j < 4; ++j) {
          v16h A = loadA(sW2, 16 * m + eL, PITCH2, 32 * j, off8);
#pragma unroll
          for (int t = 0; t < TP; ++t) {
            if (j == 0) {
              v8f z = {};
              acc[cur][t] = wmma16(A, B2[t][j], z);
            } else {
              acc[cur][t] = wmma16(A, B2[t][j], acc[cur][t]);
            }
          }
        }
        if (m > 0) consume_l2(m - 1, acc[cur ^ 1]);
      }
      consume_l2(7, acc[1]);

      if (PHASE == 3) {
#pragma unroll
        for (int t = 0; t < TP; ++t) {
          float o = dot[t] + __shfl_xor(dot[t], 16, 32); // combine both feature halves
          if (half == 0 && e[t] < E) out[e[t]] = o + b3v;
        }
      }
    }
  }

  // ---- stats epilogue: reduce 16-lane halves, LDS atomics, then global atomics ----
  if (PHASE <= 2) {
#pragma unroll
    for (int m = 0; m < 8; ++m)
#pragma unroll
      for (int r = 0; r < 8; ++r) {
        float s = S[m][r], q = Q[m][r];
        s += __shfl_xor(s, 1, 32); s += __shfl_xor(s, 2, 32);
        s += __shfl_xor(s, 4, 32); s += __shfl_xor(s, 8, 32);
        q += __shfl_xor(q, 1, 32); q += __shfl_xor(q, 2, 32);
        q += __shfl_xor(q, 4, 32); q += __shfl_xor(q, 8, 32);
        if (eL == 0) {
          int f = 16 * m + r + off8;
          atomicAdd(&sSum[f], s);
          atomicAdd(&sSq[f], q);
        }
      }
    __syncthreads();
    for (int i = tid; i < HD; i += BLOCK) {
      atomicAdd(&gsum[i], sSum[i]);
      atomicAdd(&gsq[i],  sSq[i]);
    }
  }
}

extern "C" void kernel_launch(void* const* d_in, const int* in_sizes, int n_in,
                              void* d_out, int out_size, void* d_ws, size_t ws_size,
                              hipStream_t stream) {
  const float* x   = (const float*)d_in[0];
  const int*   ei  = (const int*)d_in[1];     // edge_index [2][E]
  const float* W1  = (const float*)d_in[2];
  // d_in[3] = b1: absorbed by training-mode BN (cancels exactly)
  const float* g1  = (const float*)d_in[4];
  const float* be1 = (const float*)d_in[5];
  const float* W2  = (const float*)d_in[6];
  // d_in[7] = b2: absorbed
  const float* g2  = (const float*)d_in[8];
  const float* be2 = (const float*)d_in[9];
  const float* W3  = (const float*)d_in[10];
  const float* b3  = (const float*)d_in[11];
  float* out = (float*)d_out;

  const int E = in_sizes[1] / 2;

  float*  ws = (float*)d_ws;
  float*  s1 = ws;            // [128]
  float*  q1 = ws + 128;      // [128]
  float*  s2 = ws + 256;      // [128]
  float*  q2 = ws + 384;      // [128]
  float4* c1 = (float4*)(ws + 512);    // [128] (a, c, -, -)
  float4* c2 = (float4*)(ws + 1024);   // [128] (a, c, w3, -)

  const int nt = (E + 15) / 16;
  int blocks = (nt + 7) / 8;           // 8 waves per block
  if (blocks > 1024) blocks = 1024;
  if (blocks < 1) blocks = 1;
  const float invE = 1.0f / (float)E;

  zero_stats_kernel<<<1, 512, 0, stream>>>(ws);

  edge_mlp_kernel<1><<<blocks, BLOCK, 0, stream>>>(x, ei, W1, W2, nullptr, nullptr,
                                                   s1, q1, nullptr, nullptr, E);
  finalize_kernel<<<1, 128, 0, stream>>>(s1, q1, g1, be1, nullptr, c1, invE);

  edge_mlp_kernel<2><<<blocks, BLOCK, 0, stream>>>(x, ei, W1, W2, c1, nullptr,
                                                   s2, q2, nullptr, nullptr, E);
  finalize_kernel<<<1, 128, 0, stream>>>(s2, q2, g2, be2, W3, c2, invE);

  edge_mlp_kernel<3><<<blocks, BLOCK, 0, stream>>>(x, ei, W1, W2, c1, c2,
                                                   nullptr, nullptr, out, b3, E);
}
